// PromptEncoder_38165079392693
// MI455X (gfx1250) — matrix operand order repaired and checked by reading, not compiled, not run on hardware
//
#include <hip/hip_runtime.h>

// ---------------- problem constants (from reference setup_inputs) ----------------
#define HD   64            // hidden dim
#define K3   192           // 3*H
#define NE   65536         // num entities
#define EE   262144        // num edges
#define RRN  200           // relation_num
#define BQ   40            // batch of queries
#define RT   (RRN * BQ)    // 8000 total relations
#define LL   3             // layers
#define PH1  4             // path_hop + 1
#define SHOT 5
#define BB   (BQ / SHOT)   // 8

typedef __attribute__((ext_vector_type(16))) _Float16 v16h;
typedef __attribute__((ext_vector_type(8)))  float    v8f;

__device__ __forceinline__ float actf(float x) {
  const float s = (1.0f / 8.0f + 1.0f / 3.0f) * 0.5f; // RReLU eval slope
  return x >= 0.0f ? x : x * s;
}

__device__ __forceinline__ void cvt4(_Float16* d, float4 v) {
  d[0] = (_Float16)v.x; d[1] = (_Float16)v.y; d[2] = (_Float16)v.z; d[3] = (_Float16)v.w;
}

// A fragment (16-bit A 16x32, contiguous k order, same as the B loads): lane holds row m = lane%16, half g = lane/16;
// vector slot pair (2v,2v+1) covers K = k0 + g*16 + v*2 (+1)
__device__ __forceinline__ v16h load_afrag(const _Float16* frow, int k0, int g) {
  v16h a;
  #pragma unroll
  for (int v = 0; v < 8; ++v) {
    int K = k0 + g * 16 + v * 2;
    a[2 * v]     = frow[K];
    a[2 * v + 1] = frow[K + 1];
  }
  return a;
}

// ---------------- small utility kernels ----------------
__global__ void k_fill(float* p, float v, int n) {
  int i = blockIdx.x * blockDim.x + threadIdx.x;
  if (i < n) p[i] = v;
}

__global__ void k_init_node(const int* labels, const float* pos_emb, float* node_emb) {
  int i = blockIdx.x * blockDim.x + threadIdx.x;   // over N * (H/4) float4s
  if (i >= NE * (HD / 4)) return;
  int n = i >> 4, c4 = i & 15;
  int pos = labels[n * 2] * PH1 + labels[n * 2 + 1];
  ((float4*)node_emb)[i] = ((const float4*)(pos_emb + pos * HD))[c4];
}

__global__ void k_set_rows(float* dst, const int* idx, const float* srcRow, int cnt) {
  int i = blockIdx.x * blockDim.x + threadIdx.x;
  if (i >= cnt * HD) return;
  int b = i >> 6, h = i & 63;
  dst[idx[b] * HD + h] = srcRow[h];
}

__global__ void k_set_qrows(float* rel_emb, const int* qrel, const float* srcRow) {
  int i = blockIdx.x * blockDim.x + threadIdx.x;
  if (i >= BQ * HD) return;
  int b = i >> 6, h = i & 63;
  int r = qrel[b] + b * RRN;
  rel_emb[r * HD + h] = srcRow[h];
}

__global__ void k_loop_init(float* loopv, const float* loop_emb) {
  int i = blockIdx.x * blockDim.x + threadIdx.x;
  if (i >= BQ * HD) return;
  loopv[i] = loop_emb[i & 63];
}

__global__ void k_deg_count(const int* row, float* deg) {
  int e = blockIdx.x * blockDim.x + threadIdx.x;
  if (e < EE) atomicAdd(&deg[row[e]], 1.0f);
}

__global__ void k_deg_inv(float* deg) {
  int n = blockIdx.x * blockDim.x + threadIdx.x;
  if (n < NE) { float d = deg[n]; deg[n] = d > 0.0f ? rsqrtf(d) : 0.0f; }
}

__global__ void k_ent_norm(const int* row, const int* col, const float* dinv, float* entn) {
  int e = blockIdx.x * blockDim.x + threadIdx.x;
  if (e < EE) entn[e] = dinv[row[e]] * dinv[col[e]];
}

// per-edge attention scalar: e^(act?(cat(r_e,q_e)) . W2 + b), atomic segment sum
__global__ void k_edge_attn(const float* rel_emb, const int* etype, const int* eqr,
                            const int* sidx, const float* W2, const float* bptr,
                            float* aout, float* asum, int doAct) {
  int e = blockIdx.x * blockDim.x + threadIdx.x;
  if (e >= EE) return;
  const float4* r4 = (const float4*)(rel_emb + (size_t)etype[e] * HD);
  const float4* q4 = (const float4*)(rel_emb + (size_t)eqr[e] * HD);
  const float4* w4 = (const float4*)W2;
  float acc = bptr[0];
  #pragma unroll 4
  for (int k = 0; k < HD / 4; ++k) {
    float4 rv = r4[k], qv = q4[k], wr = w4[k], wq = w4[HD / 4 + k];
    if (doAct) {
      rv.x = actf(rv.x); rv.y = actf(rv.y); rv.z = actf(rv.z); rv.w = actf(rv.w);
      qv.x = actf(qv.x); qv.y = actf(qv.y); qv.z = actf(qv.z); qv.w = actf(qv.w);
    }
    acc += rv.x * wr.x + rv.y * wr.y + rv.z * wr.z + rv.w * wr.w;
    acc += qv.x * wq.x + qv.y * wq.y + qv.z * wq.z + qv.w * wq.w;
  }
  float ev = expf(acc);
  aout[e] = ev;
  atomicAdd(&asum[sidx[e]], ev);
}

// ---------------- big WMMA edge GEMM + scatter ----------------
// feat[e] = cat(node_emb[col], phase? node_emb[row] : rel_emb[etype], rel_emb[eqr])  [E,192]
// msg = act(feat @ W^T + bias) * scale[e];  atomic scatter-add into dst[sidx[e]]
__global__ __launch_bounds__(128)
void k_edge_gemm(const float* __restrict__ node_emb, const float* __restrict__ rel_emb,
                 const float* __restrict__ W, const float* __restrict__ bias,
                 const int* __restrict__ col, const int* __restrict__ row,
                 const int* __restrict__ etype, const int* __restrict__ eqr,
                 const float* __restrict__ escal, const float* __restrict__ ssum,
                 const float* __restrict__ entn,
                 float* __restrict__ dst, int phase) {
  __shared__ __align__(32) _Float16 Wsh[HD][K3];   // Wsh[n][k]  (B = W^T)
  __shared__ __align__(32) _Float16 Fsh[64][208];  // 64 edges x 192 (padded)
  __shared__ float Ssh[64];
  __shared__ int   Rsh[64];
  __shared__ int   Ash[64], Bsh[64], Qsh[64];
  __shared__ float biasSh[HD];

  const int tid  = threadIdx.x;
  const int lane = tid & 31;
  const int wave = tid >> 5;
  const int eb   = blockIdx.x * 64;

  if (tid < 64) {
    int e = eb + tid;
    Ash[tid] = col[e];
    Bsh[tid] = phase ? row[e] : etype[e];
    Qsh[tid] = eqr[e];
    const int* sidx = phase ? etype : row;
    int s0 = sidx[e];
    int ss = sidx[s0];                       // faithful double index_select
    float sc = escal[e] / (ssum[ss] + 1e-10f);
    if (!phase) { float nn = entn[e]; sc *= nn * nn; }  // ent_norm applied twice
    Ssh[tid] = sc;
    Rsh[tid] = s0;
  } else {
    biasSh[tid - 64] = bias[tid - 64];
  }
  // stage weights f32 -> f16 (float4 loads, contiguous)
  for (int idx = tid; idx < HD * K3 / 4; idx += 128)
    cvt4(((_Float16*)Wsh) + idx * 4, ((const float4*)W)[idx]);
  __syncthreads();

  // gather + convert edge features (each 64-float row fetched as float4s)
  const float* btab = phase ? node_emb : rel_emb;
  for (int idx = tid; idx < 64 * (K3 / 4); idx += 128) {
    int el = idx / 48;
    int c4 = idx - el * 48;
    const float* src;
    if (c4 < 16)      src = node_emb + (size_t)Ash[el] * HD + c4 * 4;
    else if (c4 < 32) src = btab     + (size_t)Bsh[el] * HD + (c4 - 16) * 4;
    else              src = rel_emb  + (size_t)Qsh[el] * HD + (c4 - 32) * 4;
    cvt4(&Fsh[el][c4 * 4], *(const float4*)src);
  }
  __syncthreads();

  const int m = lane & 15;
  const int g = lane >> 4;
  const _Float16* frow = &Fsh[wave * 16 + m][0];

  v16h af[6];
  #pragma unroll
  for (int ks = 0; ks < 6; ++ks) af[ks] = load_afrag(frow, ks * 32, g);

  #pragma unroll
  for (int nt = 0; nt < 4; ++nt) {
    const int n = nt * 16 + m;
    v8f c = {};
    #pragma unroll
    for (int ks = 0; ks < 6; ++ks) {
      const v16h b = *(const v16h*)&Wsh[n][ks * 32 + g * 16];
      c = __builtin_amdgcn_wmma_f32_16x16x32_f16(false, af[ks], false, b,
                                                 (short)0, c, false, false);
    }
    #pragma unroll
    for (int r = 0; r < 8; ++r) {
      int ml = g * 8 + r;
      float val = actf(c[r] + biasSh[n]) * Ssh[wave * 16 + ml];
      atomicAdd(&dst[(size_t)Rsh[wave * 16 + ml] * HD + n], val);
    }
  }
}

// ---------------- WMMA row GEMM + (residual) + layernorm ----------------
// out_rows = layer_norm(act(X @ W^T + bias) [+ resid]); optional dual store into out2.
// 64 contiguous rows per block, 128 threads (4 waves), KK = 64 or 192.
template <int KK>
__global__ __launch_bounds__(128)
void k_row_gemm_ln(const float* __restrict__ X, const float* __restrict__ W,
                   const float* __restrict__ bias, const float* __restrict__ resid,
                   float* __restrict__ out,
                   float* __restrict__ out2, int out2Stride, int out2Off) {
  __shared__ __align__(32) _Float16 Xsh[64][KK + 16];
  __shared__ __align__(32) _Float16 Wsh[HD][KK];
  __shared__ float Osh[64][HD + 4];
  __shared__ float muSh[64], rsSh[64];
  __shared__ float biasSh[HD];

  const int tid  = threadIdx.x;
  const int lane = tid & 31;
  const int wave = tid >> 5;
  const int rb   = blockIdx.x * 64;

  if (tid < HD) biasSh[tid] = bias[tid];
  for (int idx = tid; idx < HD * KK / 4; idx += 128)
    cvt4(((_Float16*)Wsh) + idx * 4, ((const float4*)W)[idx]);
  for (int idx = tid; idx < 64 * (KK / 4); idx += 128) {
    int r = idx / (KK / 4), c4 = idx - r * (KK / 4);
    cvt4(&Xsh[r][c4 * 4], *(const float4*)(X + (size_t)(rb + r) * KK + c4 * 4));
  }
  __syncthreads();

  const int m = lane & 15;
  const int g = lane >> 4;
  const _Float16* frow = &Xsh[wave * 16 + m][0];

  v16h af[KK / 32];
  #pragma unroll
  for (int ks = 0; ks < KK / 32; ++ks) af[ks] = load_afrag(frow, ks * 32, g);

  #pragma unroll
  for (int nt = 0; nt < 4; ++nt) {
    const int n = nt * 16 + m;
    v8f c = {};
    #pragma unroll
    for (int ks = 0; ks < KK / 32; ++ks) {
      const v16h b = *(const v16h*)&Wsh[n][ks * 32 + g * 16];
      c = __builtin_amdgcn_wmma_f32_16x16x32_f16(false, af[ks], false, b,
                                                 (short)0, c, false, false);
    }
    #pragma unroll
    for (int r = 0; r < 8; ++r) {
      int ml = g * 8 + r;            // row within this wave's 16-row tile
      int rloc = wave * 16 + ml;     // row within block tile
      float v = actf(c[r] + biasSh[n]);
      if (resid) v += resid[(size_t)(rb + rloc) * HD + n];
      Osh[rloc][n] = v;
    }
  }
  __syncthreads();

  if (tid < 64) {
    float mean = 0.f, m2 = 0.f;
    #pragma unroll 4
    for (int k = 0; k < HD; ++k) { float t = Osh[tid][k]; mean += t; m2 += t * t; }
    mean *= (1.0f / HD); m2 *= (1.0f / HD);
    muSh[tid] = mean;
    rsSh[tid] = rsqrtf((m2 - mean * mean) + 1e-5f);
  }
  __syncthreads();

  // vectorized normalized write-back (b128 stores)
  for (int idx = tid; idx < 64 * (HD / 4); idx += 128) {
    int rloc = idx >> 4, c4 = idx & 15;
    float mu = muSh[rloc], rs = rsSh[rloc];
    float4 o;
    o.x = (Osh[rloc][c4 * 4 + 0] - mu) * rs;
    o.y = (Osh[rloc][c4 * 4 + 1] - mu) * rs;
    o.z = (Osh[rloc][c4 * 4 + 2] - mu) * rs;
    o.w = (Osh[rloc][c4 * 4 + 3] - mu) * rs;
    *(float4*)(out + (size_t)(rb + rloc) * HD + c4 * 4) = o;
    if (out2) *(float4*)(out2 + (size_t)(rb + rloc) * out2Stride + out2Off + c4 * 4) = o;
  }
}

// ---------------- remaining small kernels ----------------
__global__ void k_loop_update(const float* rel_emb, const int* qrel,
                              const float* Wk /*[H][2H]*/, const float* bk, float* loopv) {
  __shared__ float cat[2 * HD];
  __shared__ float sh[HD];
  int b = blockIdx.x, h = threadIdx.x;
  int qr = qrel[b] + b * RRN;
  cat[h]      = loopv[b * HD + h];
  cat[HD + h] = rel_emb[(size_t)qr * HD + h];
  __syncthreads();
  float acc = bk[h];
  for (int k = 0; k < 2 * HD; ++k) acc += cat[k] * Wk[h * 2 * HD + k];
  float v = cat[h] + actf(acc);
  sh[h] = v;
  __syncthreads();
  float mean = 0.f, m2 = 0.f;
  for (int k = 0; k < HD; ++k) { float t = sh[k]; mean += t; m2 += t * t; }
  mean *= (1.0f / HD); m2 *= (1.0f / HD);
  loopv[b * HD + h] = (v - mean) * rsqrtf((m2 - mean * mean) + 1e-5f);
}

__global__ void k_output(const float* fin, const float* loopv, float* out) {
  int i = blockIdx.x * blockDim.x + threadIdx.x;
  if (i >= BB * (RRN + 1) * HD) return;
  int h = i & 63;
  int rb = i >> 6;
  int r = rb % (RRN + 1);
  int b = rb / (RRN + 1);
  float acc = 0.f;
  if (r < RRN) {
    for (int s = 0; s < SHOT; ++s) acc += fin[((size_t)(b * SHOT + s) * RRN + r) * HD + h];
  } else {
    for (int s = 0; s < SHOT; ++s) acc += loopv[(b * SHOT + s) * HD + h];
  }
  out[i] = acc * (1.0f / SHOT);
}

// ---------------- host launch ----------------
extern "C" void kernel_launch(void* const* d_in, const int* in_sizes, int n_in,
                              void* d_out, int out_size, void* d_ws, size_t ws_size,
                              hipStream_t stream) {
  const int* edge_index = (const int*)d_in[0];
  const int* col  = edge_index;           // edge_index[0]
  const int* row  = edge_index + EE;      // edge_index[1]
  const int* etyp = (const int*)d_in[1];
  const int* eqr  = (const int*)d_in[2];
  const int* hpos = (const int*)d_in[3];
  const int* tpos = (const int*)d_in[4];
  const int* qrel = (const int*)d_in[5];
  const int* labels = (const int*)d_in[6];
  const float* pos_emb = (const float*)d_in[10];
  const float* start_rel = (const float*)d_in[11];
  const float* loop_emb  = (const float*)d_in[12];
  const float* W_msg  = (const float*)d_in[13]; const float* b_msg  = (const float*)d_in[14];
  const float* W_ht2r = (const float*)d_in[15]; const float* b_ht2r = (const float*)d_in[16];
  const float* W_alpha= (const float*)d_in[17]; const float* b_alpha= (const float*)d_in[18];
  const float* W_beta = (const float*)d_in[19]; const float* b_beta = (const float*)d_in[20];
  const float* W_loop = (const float*)d_in[21]; const float* b_loop = (const float*)d_in[22];
  const float* W_ent  = (const float*)d_in[23]; const float* b_ent  = (const float*)d_in[24];
  const float* W_rel  = (const float*)d_in[25]; const float* b_rel  = (const float*)d_in[26];
  const float* W_final= (const float*)d_in[27]; const float* b_final= (const float*)d_in[28];
  float* out = (float*)d_out;

  // workspace carve-up (floats)
  float* ws = (float*)d_ws;
  size_t o = 0;
  float* node_emb = ws + o; o += (size_t)NE * HD;
  float* node_acc = ws + o; o += (size_t)NE * HD;
  float* rel_emb  = ws + o; o += (size_t)RT * HD;
  float* agg      = ws + o; o += (size_t)RT * HD;
  float* finals   = ws + o; o += (size_t)RT * K3;
  float* fin      = ws + o; o += (size_t)RT * HD;
  float* a_e      = ws + o; o += (size_t)EE;
  float* b_e      = ws + o; o += (size_t)EE;
  float* entn     = ws + o; o += (size_t)EE;
  float* asum     = ws + o; o += (size_t)NE;
  float* deg      = ws + o; o += (size_t)NE;
  float* bsum     = ws + o; o += (size_t)RT;
  float* loopv    = ws + o; o += (size_t)BQ * HD;

  const int T = 256;
  auto g = [](int n) { return (n + 255) / 256; };

  // ---- init ----
  k_init_node<<<g(NE * HD / 4), T, 0, stream>>>(labels, pos_emb, node_emb);
  k_set_rows<<<g(BQ * HD), T, 0, stream>>>(node_emb, hpos, pos_emb, BQ);        // pos_emb[0]
  k_set_rows<<<g(BQ * HD), T, 0, stream>>>(node_emb, tpos, pos_emb + HD, BQ);   // pos_emb[1]
  k_fill<<<g(RT * HD), T, 0, stream>>>(rel_emb, 0.f, RT * HD);
  k_set_qrows<<<g(BQ * HD), T, 0, stream>>>(rel_emb, qrel, start_rel);
  k_loop_init<<<g(BQ * HD), T, 0, stream>>>(loopv, loop_emb);
  k_fill<<<g(NE), T, 0, stream>>>(deg, 0.f, NE);
  k_deg_count<<<g(EE), T, 0, stream>>>(row, deg);
  k_deg_inv<<<g(NE), T, 0, stream>>>(deg);
  k_ent_norm<<<g(EE), T, 0, stream>>>(row, col, deg, entn);

  // ---- layers ----
  for (int i = 0; i < LL; ++i) {
    const float* Wm = W_msg  + (size_t)i * HD * K3; const float* bm = b_msg  + i * HD;
    const float* Wh = W_ht2r + (size_t)i * HD * K3; const float* bh = b_ht2r + i * HD;
    const float* Wa = W_alpha + (size_t)i * 2 * HD; const float* ba = b_alpha + i;
    const float* Wb = W_beta  + (size_t)i * 2 * HD; const float* bb = b_beta  + i;
    const float* Wl = W_loop + (size_t)i * HD * 2 * HD; const float* bl = b_loop + i * HD;
    const float* We = W_ent  + (size_t)i * HD * HD; const float* be = b_ent + i * HD;
    const float* Wr = W_rel  + (size_t)i * HD * HD; const float* br = b_rel + i * HD;

    // entity message pass
    k_fill<<<g(NE), T, 0, stream>>>(asum, 0.f, NE);
    k_fill<<<g(NE * HD), T, 0, stream>>>(node_acc, 0.f, NE * HD);
    k_edge_attn<<<g(EE), T, 0, stream>>>(rel_emb, etyp, eqr, row, Wa, ba, a_e, asum, 1);
    k_edge_gemm<<<EE / 64, 128, 0, stream>>>(node_emb, rel_emb, Wm, bm,
                                             col, row, etyp, eqr,
                                             a_e, asum, entn, node_acc, 0);
    // node_emb = LN(act(node_acc @ W_ent^T + b_ent))   [WMMA]
    k_row_gemm_ln<HD><<<NE / 64, 128, 0, stream>>>(node_acc, We, be, nullptr,
                                                   node_emb, nullptr, 0, 0);

    // relation message pass
    k_fill<<<g(RT), T, 0, stream>>>(bsum, 0.f, RT);
    k_fill<<<g(RT * HD), T, 0, stream>>>(agg, 0.f, RT * HD);
    k_edge_attn<<<g(EE), T, 0, stream>>>(rel_emb, etyp, eqr, etyp, Wb, bb, b_e, bsum, 0);
    k_edge_gemm<<<EE / 64, 128, 0, stream>>>(node_emb, rel_emb, Wh, bh,
                                             col, row, etyp, eqr,
                                             b_e, bsum, nullptr, agg, 1);
    // rel_emb = LN(act(agg @ W_rel^T + b_rel) + rel_emb); finals[:, i*H:(i+1)*H]   [WMMA]
    k_row_gemm_ln<HD><<<RT / 64, 128, 0, stream>>>(agg, Wr, br, rel_emb,
                                                   rel_emb, finals, K3, i * HD);
    k_loop_update<<<BQ, HD, 0, stream>>>(rel_emb, qrel, Wl, bl, loopv);
  }

  // ---- final projection: fin = LN(act(finals @ W_final^T + b_final))   [WMMA] ----
  k_row_gemm_ln<K3><<<RT / 64, 128, 0, stream>>>(finals, W_final, b_final, nullptr,
                                                 fin, nullptr, 0, 0);
  k_output<<<g(BB * (RRN + 1) * HD), T, 0, stream>>>(fin, loopv, out);
}
